// MolEncoder_29901562314931
// MI455X (gfx1250) — matrix operand order repaired and checked by reading, not compiled, not run on hardware
//
#include <hip/hip_runtime.h>
#include <hip/hip_bf16.h>
#include <math.h>

// ---------------------------------------------------------------------------
// Mamba forward for gfx1250 (MI455X).
// GEMMs: double-buffered LDS-staged WMMA (v_wmma_f32_16x16x32_f16),
//        128x64 block tile, 32x32 per wave (2x2 fragments, 4 WMMA/K-step).
// SSM scan: LDS-staged sequential kernel, one thread per (b, d_inner) channel.
// ---------------------------------------------------------------------------

typedef _Float16 half_t;
typedef __attribute__((ext_vector_type(2)))  _Float16 v2h;
typedef __attribute__((ext_vector_type(16))) _Float16 v16h;
typedef __attribute__((ext_vector_type(8)))  float    v8f;

#define DM 512
#define DI 1024
#define DS 16
#define DR 32
#define NL 8
#define VOC 2048
#define BB 4
#define LL 1024

// ------------------------------ GEMM (WMMA) --------------------------------
// C[M,N] = A[M,K] * W^T, W stored row-major (N,K) (jax/torch weight layout).
// Block: 256 threads (8 waves) -> 128x64 C tile. Wave (wm,wn) -> 32x32.
// Requires: M % 128 == 0, N % 64 == 0, K % 32 == 0, 16B-aligned A/W rows.
__global__ void __launch_bounds__(256)
gemm_wmma_f16_kernel(const half_t* __restrict__ A, const half_t* __restrict__ W,
                     float* __restrict__ C, int M, int N, int K)
{
    __shared__ half_t lsA[2][128 * 32];   // 2 x 8KB
    __shared__ half_t lsW[2][64 * 32];    // 2 x 4KB

    const int tid  = threadIdx.x;
    const int lane = tid & 31;
    const int wave = tid >> 5;
    const int wm   = wave & 3;            // 0..3: 32-row band
    const int wn   = wave >> 2;           // 0..1: 32-col band
    const int hl   = lane >> 4;           // half-wave select (ISA 16-bit layouts)
    const int mr   = lane & 15;

    const int tile_m = blockIdx.y * 128;
    const int tile_n = blockIdx.x * 64;

    // cooperative staging: A = 512 uint4 chunks (4/row), W = 256 uint4 chunks
    const int srow = tid >> 2;            // 0..63
    const int sseg = (tid & 3) * 8;       // k-offset in halves
    const half_t* __restrict__ Ag0 = A + (size_t)(tile_m + srow) * K + sseg;
    const half_t* __restrict__ Ag1 = A + (size_t)(tile_m + 64 + srow) * K + sseg;
    const half_t* __restrict__ Wg  = W + (size_t)(tile_n + srow) * K + sseg;

    uint4 ra0 = *(const uint4*)Ag0;
    uint4 ra1 = *(const uint4*)Ag1;
    uint4 rw  = *(const uint4*)Wg;

    v8f acc[2][2] = {};
    const int ksteps = K >> 5;
    int buf = 0;

    ((uint4*)lsA[0])[tid]       = ra0;
    ((uint4*)lsA[0])[tid + 256] = ra1;
    ((uint4*)lsW[0])[tid]       = rw;
    __syncthreads();

    for (int kt = 0; kt < ksteps; ++kt) {
        const int nxt = buf ^ 1;
        const bool more = (kt + 1 < ksteps);
        if (more) {
            const int ko = (kt + 1) << 5;
            __builtin_prefetch(Ag0 + ko + 32, 0, 1);   // global_prefetch_b8
            ra0 = *(const uint4*)(Ag0 + ko);
            ra1 = *(const uint4*)(Ag1 + ko);
            rw  = *(const uint4*)(Wg + ko);
        }

        // A fragments (16x32 f16): lane holds row m; 16 contiguous halves,
        // lanes 16-31 cover K=16..31 (matches B)
        v16h afr[2], bfr[2];
#pragma unroll
        for (int f = 0; f < 2; ++f) {
            const int row = wm * 32 + f * 16 + mr;
            afr[f] = *(const v16h*)&lsA[buf][row * 32 + 16 * hl];
        }
        // B fragments (32x16): lane holds column n (= row of W); 16 contiguous
        // halves, lanes 16-31 cover K=16..31 -> single 32B LDS read.
#pragma unroll
        for (int g = 0; g < 2; ++g) {
            const int n = wn * 32 + g * 16 + mr;
            bfr[g] = *(const v16h*)&lsW[buf][n * 32 + 16 * hl];
        }

#pragma unroll
        for (int f = 0; f < 2; ++f)
#pragma unroll
            for (int g = 0; g < 2; ++g)
                acc[f][g] = __builtin_amdgcn_wmma_f32_16x16x32_f16(
                    false, afr[f], false, bfr[g], (short)0, acc[f][g],
                    false, false);

        if (more) {
            ((uint4*)lsA[nxt])[tid]       = ra0;
            ((uint4*)lsA[nxt])[tid + 256] = ra1;
            ((uint4*)lsW[nxt])[tid]       = rw;
        }
        __syncthreads();
        buf = nxt;
    }

    // C/D layout: VGPR r -> row (r + 8*hl), col = lane&15
#pragma unroll
    for (int f = 0; f < 2; ++f)
#pragma unroll
        for (int g = 0; g < 2; ++g)
#pragma unroll
            for (int r = 0; r < 8; ++r)
                C[(size_t)(tile_m + wm * 32 + f * 16 + r + 8 * hl) * N
                  + tile_n + wn * 32 + g * 16 + mr] = acc[f][g][r];
    (void)M;
}

// --------------------------- elementwise helpers ---------------------------
__device__ __forceinline__ float siluf(float x) { return x / (1.f + __expf(-x)); }

__global__ void embed_kernel(const int* __restrict__ idx, const float* __restrict__ emb,
                             float* __restrict__ hid, int n)
{
    int i = blockIdx.x * 256 + threadIdx.x;
    if (i >= n) return;
    int row = i >> 9, c = i & 511;
    hid[i] = emb[(size_t)idx[row] * DM + c];
}

// mode 0: resid += hid, LN(resid) | mode 1: resid = hid, LN(resid)
// mode 2: plain LN(hid) (resid untouched)
__global__ void __launch_bounds__(256)
add_ln_kernel(const float* __restrict__ hid, float* __restrict__ resid,
              float* __restrict__ out, const float* __restrict__ w,
              const float* __restrict__ b, int mode)
{
    __shared__ float rowb[DM];
    __shared__ float red[256];
    const int row = blockIdx.x, tid = threadIdx.x;
    const size_t base = (size_t)row * DM;
    float s = 0.f;
    for (int d = tid; d < DM; d += 256) {
        float v = hid[base + d];
        if (mode == 0) v += resid[base + d];
        if (mode != 2) resid[base + d] = v;
        rowb[d] = v; s += v;
    }
    red[tid] = s; __syncthreads();
    for (int off = 128; off > 0; off >>= 1) { if (tid < off) red[tid] += red[tid + off]; __syncthreads(); }
    const float mean = red[0] * (1.f / DM);
    __syncthreads();
    float s2 = 0.f;
    for (int d = tid; d < DM; d += 256) { float dv = rowb[d] - mean; s2 += dv * dv; }
    red[tid] = s2; __syncthreads();
    for (int off = 128; off > 0; off >>= 1) { if (tid < off) red[tid] += red[tid + off]; __syncthreads(); }
    const float rstd = rsqrtf(red[0] * (1.f / DM) + 1e-5f);
    __syncthreads();
    for (int d = tid; d < DM; d += 256)
        out[base + d] = (rowb[d] - mean) * rstd * w[d] + b[d];
}

__global__ void convert_f16_kernel(const float* __restrict__ s, half_t* __restrict__ d, size_t n)
{
    size_t i = (size_t)blockIdx.x * 256 + threadIdx.x;
    if (i < n) d[i] = (half_t)s[i];
}

__global__ void convert_f16_strided_kernel(const float* __restrict__ s, half_t* __restrict__ d,
                                           int rows, int cols, int stride)
{
    int i = blockIdx.x * 256 + threadIdx.x;
    if (i >= rows * cols) return;
    int r = i / cols, c = i - r * cols;
    d[i] = (half_t)s[(size_t)r * stride + c];
}

// causal depthwise conv (width 4) over time + SiLU; input is first half of xz
__global__ void conv_silu_kernel(const float* __restrict__ xz, const float* __restrict__ cw,
                                 const float* __restrict__ cb, float* __restrict__ xi, int n)
{
    int i = blockIdx.x * 256 + threadIdx.x;
    if (i >= n) return;
    int d = i & (DI - 1);
    size_t row = (size_t)i >> 10;           // b*L + l
    int l = (int)(row & (LL - 1));
    size_t rb = row - l;                    // b*L
    float acc = cb[d];
#pragma unroll
    for (int j = 0; j < 4; ++j) {
        int ll = l - 3 + j;
        if (ll >= 0) acc += cw[d * 4 + j] * xz[(rb + ll) * (2 * DI) + d];
    }
    xi[i] = siluf(acc);
}

__global__ void softplus_bias_kernel(float* __restrict__ delta, const float* __restrict__ bias, int n)
{
    int i = blockIdx.x * 256 + threadIdx.x;
    if (i >= n) return;
    float v = delta[i] + bias[i & (DI - 1)];
    delta[i] = (v > 20.f) ? v : log1pf(__expf(v));
}

// selective scan: one thread per (b,d); B/C vectors staged in LDS per 64 steps
#define TS 64
__global__ void __launch_bounds__(256)
ssm_scan_kernel(const float* __restrict__ delta, const float* __restrict__ xi,
                const float* __restrict__ xdbl, const float* __restrict__ A_log,
                float* __restrict__ ys)
{
    __shared__ float bc[TS * 32];
    const int tid = threadIdx.x;
    const int d = blockIdx.x * 256 + tid;
    const int b = blockIdx.y;
    float A[DS], h[DS];
#pragma unroll
    for (int n = 0; n < DS; ++n) { A[n] = -__expf(A_log[d * DS + n]); h[n] = 0.f; }
    for (int lc = 0; lc < LL; lc += TS) {
        __syncthreads();
        for (int i = tid; i < TS * 32; i += 256) {
            int t = i >> 5, j = i & 31;
            bc[i] = xdbl[((size_t)b * LL + lc + t) * 64 + 32 + j];   // [B(16) | C(16)]
        }
        __syncthreads();
        for (int t = 0; t < TS; ++t) {
            const size_t r = (size_t)b * LL + lc + t;
            const float dlt = delta[r * DI + d];
            const float dx  = dlt * xi[r * DI + d];
            float y = 0.f;
#pragma unroll
            for (int n = 0; n < DS; ++n) {
                h[n] = __expf(dlt * A[n]) * h[n] + dx * bc[t * 32 + n];
                y += h[n] * bc[t * 32 + 16 + n];
            }
            ys[r * DI + d] = y;
        }
    }
}

__global__ void gate_kernel(const float* __restrict__ ys, const float* __restrict__ xi,
                            const float* __restrict__ Dp, const float* __restrict__ xz,
                            float* __restrict__ yg, int n)
{
    int i = blockIdx.x * 256 + threadIdx.x;
    if (i >= n) return;
    int d = i & (DI - 1);
    size_t row = (size_t)i >> 10;
    float z = xz[row * (2 * DI) + DI + d];
    yg[i] = (ys[i] + xi[i] * Dp[d]) * siluf(z);
}

__global__ void bias_gelu_kernel(float* __restrict__ x, const float* __restrict__ bias, int n)
{
    int i = blockIdx.x * 256 + threadIdx.x;
    if (i >= n) return;
    float v = x[i] + bias[i & (DM - 1)];
    x[i] = 0.5f * v * (1.f + erff(v * 0.70710678118654752f));
}

// ------------------------------- host driver -------------------------------
extern "C" void kernel_launch(void* const* d_in, const int* in_sizes, int n_in,
                              void* d_out, int out_size, void* d_ws, size_t ws_size,
                              hipStream_t stream)
{
    const int*   idx       = (const int*)  d_in[0];
    const float* tok_emb   = (const float*)d_in[1];
    const float* norm_w    = (const float*)d_in[2];
    const float* norm_b    = (const float*)d_in[3];
    const float* in_proj_w = (const float*)d_in[4];
    const float* conv_w    = (const float*)d_in[5];
    const float* conv_b    = (const float*)d_in[6];
    const float* x_proj_w  = (const float*)d_in[7];
    const float* dt_proj_w = (const float*)d_in[8];
    const float* dt_proj_b = (const float*)d_in[9];
    const float* A_log     = (const float*)d_in[10];
    const float* D_skip    = (const float*)d_in[11];
    const float* out_proj_w= (const float*)d_in[12];
    const float* norm_f_w  = (const float*)d_in[13];
    const float* norm_f_b  = (const float*)d_in[14];
    const float* le_w      = (const float*)d_in[15];
    const float* le_b      = (const float*)d_in[16];
    const float* ln_w      = (const float*)d_in[17];
    const float* ln_b      = (const float*)d_in[18];
    const float* head_w    = (const float*)d_in[19];
    (void)in_sizes; (void)n_in; (void)out_size; (void)ws_size;

    const int rows = BB * LL;   // 4096

    char* ws = (char*)d_ws;
    size_t off = 0;
    auto carve = [&](size_t bytes) -> void* {
        void* p = ws + off; off += (bytes + 255) & ~(size_t)255; return p;
    };
    float*  resid = (float*) carve((size_t)rows * DM * 4);
    float*  hid   = (float*) carve((size_t)rows * DM * 4);
    float*  normed= (float*) carve((size_t)rows * DM * 4);
    float*  xz    = (float*) carve((size_t)rows * 2 * DI * 4);
    float*  xi    = (float*) carve((size_t)rows * DI * 4);
    float*  xdbl  = (float*) carve((size_t)rows * 64 * 4);
    float*  delta = (float*) carve((size_t)rows * DI * 4);
    float*  ys    = (float*) carve((size_t)rows * DI * 4);
    float*  yg    = (float*) carve((size_t)rows * DI * 4);
    float*  tmp   = (float*) carve((size_t)rows * DM * 4);
    half_t* a16   = (half_t*)carve((size_t)rows * DI * 2);
    half_t* w16   = (half_t*)carve((size_t)2 * DI * DM * 2);

    auto cvt = [&](const float* s, half_t* d, size_t n) {
        convert_f16_kernel<<<(unsigned)((n + 255) / 256), 256, 0, stream>>>(s, d, n);
    };
    auto gemm = [&](const half_t* A, const half_t* W, float* C, int M, int N, int K) {
        dim3 g((unsigned)(N / 64), (unsigned)(M / 128));
        gemm_wmma_f16_kernel<<<g, 256, 0, stream>>>(A, W, C, M, N, K);
    };

    // embedding
    embed_kernel<<<rows * DM / 256, 256, 0, stream>>>(idx, tok_emb, hid, rows * DM);

    for (int layer = 0; layer < NL; ++layer) {
        // residual add + layernorm
        add_ln_kernel<<<rows, 256, 0, stream>>>(hid, resid, normed,
                                                norm_w + layer * DM, norm_b + layer * DM,
                                                layer == 0 ? 1 : 0);
        // in_proj: (rows,512) x (2048,512)^T -> xz
        cvt(normed, a16, (size_t)rows * DM);
        cvt(in_proj_w + (size_t)layer * 2 * DI * DM, w16, (size_t)2 * DI * DM);
        gemm(a16, w16, xz, rows, 2 * DI, DM);
        // causal depthwise conv + silu
        conv_silu_kernel<<<rows * DI / 256, 256, 0, stream>>>(
            xz, conv_w + (size_t)layer * DI * 4, conv_b + layer * DI, xi, rows * DI);
        // x_proj: (rows,1024) x (64,1024)^T -> xdbl
        cvt(xi, a16, (size_t)rows * DI);
        cvt(x_proj_w + (size_t)layer * 64 * DI, w16, (size_t)64 * DI);
        gemm(a16, w16, xdbl, rows, 64, DI);
        // dt_proj: (rows,32) x (1024,32)^T -> delta (pre-bias)
        convert_f16_strided_kernel<<<rows * DR / 256, 256, 0, stream>>>(xdbl, a16, rows, DR, 64);
        cvt(dt_proj_w + (size_t)layer * DI * DR, w16, (size_t)DI * DR);
        gemm(a16, w16, delta, rows, DI, DR);
        softplus_bias_kernel<<<rows * DI / 256, 256, 0, stream>>>(
            delta, dt_proj_b + layer * DI, rows * DI);
        // selective scan
        ssm_scan_kernel<<<dim3(DI / 256, BB), 256, 0, stream>>>(
            delta, xi, xdbl, A_log + (size_t)layer * DI * DS, ys);
        // D-skip + gate
        gate_kernel<<<rows * DI / 256, 256, 0, stream>>>(
            ys, xi, D_skip + layer * DI, xz, yg, rows * DI);
        // out_proj: (rows,1024) x (512,1024)^T -> hid
        cvt(yg, a16, (size_t)rows * DI);
        cvt(out_proj_w + (size_t)layer * DM * DI, w16, (size_t)DM * DI);
        gemm(a16, w16, hid, rows, DM, DI);
    }

    // final residual + norm_f
    add_ln_kernel<<<rows, 256, 0, stream>>>(hid, resid, normed, norm_f_w, norm_f_b, 0);
    // LangLayer: linear + bias + exact gelu
    cvt(normed, a16, (size_t)rows * DM);
    cvt(le_w, w16, (size_t)DM * DM);
    gemm(a16, w16, tmp, rows, DM, DM);
    bias_gelu_kernel<<<rows * DM / 256, 256, 0, stream>>>(tmp, le_b, rows * DM);
    // layernorm (plain)
    add_ln_kernel<<<rows, 256, 0, stream>>>(tmp, resid, normed, ln_w, ln_b, 2);
    // head: (rows,512) x (2048,512)^T -> out
    cvt(normed, a16, (size_t)rows * DM);
    cvt(head_w, w16, (size_t)VOC * DM);
    gemm(a16, w16, (float*)d_out, rows, VOC, DM);
}
